// MoDL_70497593196699
// MI455X (gfx1250) — compile-verified, matrix operand order checked by reading, not run on hardware
//
#include <hip/hip_runtime.h>
#include <hip/hip_bf16.h>
#include <math.h>

#define BN 4
#define CN 8
#define HN 384
#define WN 384
#define HWN (HN*WN)
#define NLAYER 5
#define CGITER 10
#define NCH 64
#define WREP_LAYER (9*2*64*32)   // 36864 halves per mid-conv layer

typedef __attribute__((ext_vector_type(16))) _Float16 v16h;
typedef __attribute__((ext_vector_type(8)))  float    v8f;
typedef __attribute__((ext_vector_type(2)))  float    v2f;

__device__ __forceinline__ v8f wmma_f16f32(v16h a, v16h b, v8f c) {
  return __builtin_amdgcn_wmma_f32_16x16x32_f16(false, a, false, b, (short)0, c, false, false);
}
__device__ __forceinline__ v8f wmma_f32x4(v2f a, v2f b, v8f c) {
  return __builtin_amdgcn_wmma_f32_16x16x4_f32(false, a, false, b, (short)0, c, false, false);
}

// async 16B global->LDS copy (CDNA5 TDM-lite path, ASYNCcnt tracked)
__device__ __forceinline__ void async_copy_b128(unsigned int ldsOff, const void* gptr) {
  asm volatile("global_load_async_to_lds_b128 %0, %1, off"
               :: "v"(ldsOff), "v"((unsigned long long)(uintptr_t)gptr)
               : "memory");
}
__device__ __forceinline__ void async_wait0() {
  asm volatile("s_wait_asynccnt 0x0" ::: "memory");
}

// ---------------- DFT tables: Wr = cos(-2pi jk/N), WiF = sin(-2pi jk/N), WiI = -WiF
__global__ void k_init_dft(float* __restrict__ Wr, float* __restrict__ WiF, float* __restrict__ WiI) {
  int idx = blockIdx.x * 256 + threadIdx.x;
  if (idx >= HWN) return;
  int j = idx / WN, k = idx % WN;
  int m = (j * k) % WN;
  float ang = -2.0f * 3.14159265358979323846f * (float)m / (float)WN;
  float s, c;
  sincosf(ang, &s, &c);
  Wr[idx] = c; WiF[idx] = s; WiI[idx] = -s;
}

// ---------------- repack mid weights [7][64][64][3][3] f32 -> [L][tap][half][cout][32] f16
__global__ void k_repack(const float* __restrict__ wmid, _Float16* __restrict__ wrep) {
  int idx = blockIdx.x * 256 + threadIdx.x;
  if (idx >= 7 * WREP_LAYER) return;
  int L = idx / WREP_LAYER;
  int r = idx % WREP_LAYER;
  int kk = r % 32; r /= 32;
  int cout = r % 64; r /= 64;
  int half = r % 2;
  int tap = r / 2;
  int cin = half * 32 + kk;
  int ty = tap / 3, tx = tap % 3;
  size_t src = ((((size_t)L * 64 + cout) * 64 + cin) * 3 + ty) * 3 + tx;
  wrep[idx] = (_Float16)wmid[src];
}

// ---------------- conv 2 -> 64 (scalar; tiny K), writes channels-last f16 activation
__global__ void k_conv_in(const float* __restrict__ x, const float* __restrict__ w_in,
                          const float* __restrict__ b_in, _Float16* __restrict__ act) {
  __shared__ float ws[64 * 18];
  for (int i = threadIdx.x; i < 64 * 18; i += 256) ws[i] = w_in[i];
  __syncthreads();
  int idx = blockIdx.x * 256 + threadIdx.x;
  if (idx >= BN * HWN) return;
  int b = idx / HWN, hw = idx % HWN, h = hw / WN, w = hw % WN;
  float vals[18];
#pragma unroll
  for (int cin = 0; cin < 2; ++cin)
#pragma unroll
    for (int tap = 0; tap < 9; ++tap) {
      int iy = h + tap / 3 - 1, ix = w + tap % 3 - 1;
      float v = 0.f;
      if (iy >= 0 && iy < HN && ix >= 0 && ix < WN)
        v = x[(size_t)(b * 2 + cin) * HWN + iy * WN + ix];
      vals[cin * 9 + tap] = v;
    }
  _Float16* ap = act + (size_t)idx * NCH;
  for (int co = 0; co < NCH; ++co) {
    float acc = b_in[co];
#pragma unroll
    for (int t = 0; t < 18; ++t) acc += vals[t] * ws[co * 18 + t];
    ap[co] = (_Float16)acc;
  }
}

// ---------------- conv 64 -> 64 via WMMA implicit GEMM (block: 64 px x 64 cout, K = 576)
// LDS staging done with async global->LDS copies (ASYNCcnt), halo lanes zero-filled via DS.
__global__ void __launch_bounds__(256) k_conv_mid(const _Float16* __restrict__ actIn,
                                                  const _Float16* __restrict__ wrep,
                                                  const float* __restrict__ bias,
                                                  int relu, _Float16* __restrict__ actOut) {
  __shared__ __attribute__((aligned(16))) _Float16 Alds[64][40];
  __shared__ __attribute__((aligned(16))) _Float16 Blds[64][40];
  const int tid = threadIdx.x;
  const int wave = tid >> 5, lane = tid & 31;
  const int b = blockIdx.z, h = blockIdx.y, wt = blockIdx.x;
  const int m0 = (wave >> 2) * 16, n0 = (wave & 3) * 16;
  const int lp = tid >> 2, lj = (tid & 3) * 8;   // cooperative loader: row lp, 8-half chunk lj
  const int row = lane & 15;
  const int koff = (lane >> 4) * 8;
  const unsigned int aLds = (unsigned int)(uintptr_t)(&Alds[lp][lj]);
  const unsigned int bLds = (unsigned int)(uintptr_t)(&Blds[lp][lj]);
  v8f c0 = {}, c1 = {};
  for (int tap = 0; tap < 9; ++tap) {
    int iy = h + tap / 3 - 1;
    int ix = wt * 64 + lp + (tap % 3) - 1;
    bool inb = (iy >= 0 && iy < HN && ix >= 0 && ix < WN);
    const _Float16* src = inb ? (actIn + ((size_t)b * HN + iy) * WN * NCH + (size_t)ix * NCH)
                              : (const _Float16*)0;
    // prefetch next tap's activation row into cache hierarchy
    if (tap < 8) {
      int iy2 = h + (tap + 1) / 3 - 1;
      int ix2 = wt * 64 + lp + (tap + 1) % 3 - 1;
      if (iy2 >= 0 && iy2 < HN && ix2 >= 0 && ix2 < WN)
        __builtin_prefetch(actIn + ((size_t)b * HN + iy2) * WN * NCH + (size_t)ix2 * NCH, 0, 1);
    }
    for (int half = 0; half < 2; ++half) {
      __syncthreads();
      if (inb) {
        async_copy_b128(aLds, src + half * 32 + lj);
      } else {
        *(uint4*)(&Alds[lp][lj]) = make_uint4(0u, 0u, 0u, 0u);
      }
      async_copy_b128(bLds, wrep + (((size_t)tap * 2 + half) * 64 + lp) * 32 + lj);
      asm volatile("" ::: "memory");
      async_wait0();
      __syncthreads();
      union { v16h h; unsigned int u[8]; } a0, a1, bf;
#pragma unroll
      for (int v = 0; v < 8; ++v) {
        int kb = ((v & 4) ? 16 : 0) + koff + 2 * (v & 3);   // ISA 16-bit A 16x32 layout
        a0.u[v] = *(const unsigned int*)(&Alds[m0 + row][kb]);
        a1.u[v] = *(const unsigned int*)(&Alds[m0 + 32 + row][kb]);
        bf.u[v] = *(const unsigned int*)(&Blds[n0 + row][kb]);
      }
      c0 = wmma_f16f32(a0.h, bf.h, c0);
      c1 = wmma_f16f32(a1.h, bf.h, c1);
    }
  }
  const int ncol = n0 + row;
  const float bb = bias[ncol];
  const int radd = (lane >= 16) ? 8 : 0;
#pragma unroll
  for (int r = 0; r < 8; ++r) {
    float f0 = c0[r] + bb, f1 = c1[r] + bb;
    if (relu) { f0 = fmaxf(f0, 0.f); f1 = fmaxf(f1, 0.f); }
    int p0 = wt * 64 + m0 + radd + r;
    actOut[(((size_t)b * HN + h) * WN + p0) * NCH + ncol] = (_Float16)f0;
    actOut[(((size_t)b * HN + h) * WN + p0 + 32) * NCH + ncol] = (_Float16)f1;
  }
}

// ---------------- conv 64 -> 2 + residual + (under + lam * dwout) fused, writes rhs planes
__global__ void k_conv_out(const _Float16* __restrict__ act, const float* __restrict__ w_out,
                           const float* __restrict__ b_out, const float* __restrict__ xin,
                           const float* __restrict__ under, const float* __restrict__ lamp,
                           float* __restrict__ rhs) {
  __shared__ float ws[2 * 64 * 9];
  for (int i = threadIdx.x; i < 2 * 64 * 9; i += 256) ws[i] = w_out[i];
  __syncthreads();
  int idx = blockIdx.x * 256 + threadIdx.x;
  if (idx >= BN * HWN) return;
  int b = idx / HWN, hw = idx % HWN, h = hw / WN, w = hw % WN;
  float acc0 = b_out[0], acc1 = b_out[1];
  for (int tap = 0; tap < 9; ++tap) {
    int iy = h + tap / 3 - 1, ix = w + tap % 3 - 1;
    if (iy < 0 || iy >= HN || ix < 0 || ix >= WN) continue;
    const _Float16* ap = act + (((size_t)b * HN + iy) * WN + ix) * NCH;
#pragma unroll
    for (int c = 0; c < NCH; ++c) {
      float v = (float)ap[c];
      acc0 += v * ws[c * 9 + tap];
      acc1 += v * ws[(64 + c) * 9 + tap];
    }
  }
  float lam = lamp[0];
  size_t i0 = (size_t)(b * 2) * HWN + hw, i1 = (size_t)(b * 2 + 1) * HWN + hw;
  rhs[i0] = under[i0] + lam * (xin[i0] + acc0);
  rhs[i1] = under[i1] + lam * (xin[i1] + acc1);
}

// ---------------- coil = csm * p (complex), plane layout [b][c][re/im][HW]
__global__ void k_coil_expand(const float* __restrict__ p, const float* __restrict__ csm,
                              float* __restrict__ coil) {
  int idx = blockIdx.x * 256 + threadIdx.x;
  if (idx >= BN * CN * HWN) return;
  int hw = idx % HWN;
  int bc = idx / HWN;
  int b = bc / CN;
  float cr = csm[(size_t)idx * 2], ci = csm[(size_t)idx * 2 + 1];
  float pr = p[(size_t)(b * 2) * HWN + hw], pi = p[(size_t)(b * 2 + 1) * HWN + hw];
  coil[((size_t)bc * 2) * HWN + hw]     = cr * pr - ci * pi;
  coil[((size_t)bc * 2 + 1) * HWN + hw] = cr * pi + ci * pr;
}

// ---------------- batched complex GEMM stage: Cstore[col][row] = (A * W)[row][col]
// W is the symmetric DFT table, so B fragments read contiguously along W's rows.
// Applying this kernel twice == fft2 (or ifft2 with the conjugate table + scale).
__global__ void __launch_bounds__(128) k_cgemm(
    const float* __restrict__ Ar, const float* __restrict__ Ai, long long aStride,
    const float* __restrict__ Wtr, const float* __restrict__ Wti,
    float* __restrict__ Cr, float* __restrict__ Ci, long long cStride, float scale) {
  int img = blockIdx.z;
  Ar += (size_t)img * aStride; Ai += (size_t)img * aStride;
  Cr += (size_t)img * cStride; Ci += (size_t)img * cStride;
  int wave = threadIdx.x >> 5, lane = threadIdx.x & 31;
  int r0 = blockIdx.y * 16;
  int c0 = (blockIdx.x * 4 + wave) * 16;
  int M = r0 + (lane & 15);
  int N = c0 + (lane & 15);
  int kb = (lane >> 4) * 2;   // ISA 32-bit A 16x4 layout: lanes16-31 hold K=2,3
  const float* aR = Ar + (size_t)M * WN + kb;
  const float* aI = Ai + (size_t)M * WN + kb;
  const float* bR = Wtr + (size_t)N * WN + kb;   // W[k][N] == W[N][k]
  const float* bI = Wti + (size_t)N * WN + kb;
  v8f arr = {}, aii = {}, ari = {}, air = {};
  for (int kk = 0; kk < WN; kk += 4) {
    v2f a_r = *(const v2f*)(aR + kk);
    v2f a_i = *(const v2f*)(aI + kk);
    v2f b_r = *(const v2f*)(bR + kk);
    v2f b_i = *(const v2f*)(bI + kk);
    arr = wmma_f32x4(a_r, b_r, arr);
    aii = wmma_f32x4(a_i, b_i, aii);
    ari = wmma_f32x4(a_r, b_i, ari);
    air = wmma_f32x4(a_i, b_r, air);
  }
  // transposed store: per lane 8 consecutive floats -> two b128 stores per plane
  int col = c0 + (lane & 15);
  int rb = r0 + ((lane >= 16) ? 8 : 0);
  float4 s0, s1, t0, t1;
  s0.x = (arr[0] - aii[0]) * scale; s0.y = (arr[1] - aii[1]) * scale;
  s0.z = (arr[2] - aii[2]) * scale; s0.w = (arr[3] - aii[3]) * scale;
  s1.x = (arr[4] - aii[4]) * scale; s1.y = (arr[5] - aii[5]) * scale;
  s1.z = (arr[6] - aii[6]) * scale; s1.w = (arr[7] - aii[7]) * scale;
  t0.x = (ari[0] + air[0]) * scale; t0.y = (ari[1] + air[1]) * scale;
  t0.z = (ari[2] + air[2]) * scale; t0.w = (ari[3] + air[3]) * scale;
  t1.x = (ari[4] + air[4]) * scale; t1.y = (ari[5] + air[5]) * scale;
  t1.z = (ari[6] + air[6]) * scale; t1.w = (ari[7] + air[7]) * scale;
  *(float4*)(Cr + (size_t)col * WN + rb)     = s0;
  *(float4*)(Cr + (size_t)col * WN + rb + 4) = s1;
  *(float4*)(Ci + (size_t)col * WN + rb)     = t0;
  *(float4*)(Ci + (size_t)col * WN + rb + 4) = t1;
}

// ---------------- k-space mask (broadcast over coil + re/im)
__global__ void k_mask_mul(float* __restrict__ coil, const float* __restrict__ mask) {
  int idx = blockIdx.x * 256 + threadIdx.x;
  if (idx >= BN * CN * 2 * HWN) return;
  int hw = idx % HWN;
  int b = idx / (CN * 2 * HWN);
  coil[idx] *= mask[(size_t)b * HWN + hw];
}

// ---------------- Ap = sum_c conj(csm)*img + lam*p
__global__ void k_coil_reduce(const float* __restrict__ coil, const float* __restrict__ csm,
                              const float* __restrict__ p, const float* __restrict__ lamp,
                              float* __restrict__ Ap) {
  int idx = blockIdx.x * 256 + threadIdx.x;
  if (idx >= BN * HWN) return;
  int b = idx / HWN, hw = idx % HWN;
  float sr = 0.f, si = 0.f;
#pragma unroll
  for (int c = 0; c < CN; ++c) {
    size_t bc = (size_t)b * CN + c;
    float ir = coil[bc * 2 * HWN + hw], ii = coil[(bc * 2 + 1) * HWN + hw];
    float cr = csm[(bc * HWN + hw) * 2], ci = csm[(bc * HWN + hw) * 2 + 1];
    sr += cr * ir + ci * ii;
    si += cr * ii - ci * ir;
  }
  float lam = lamp[0];
  Ap[(size_t)(b * 2) * HWN + hw]     = sr + lam * p[(size_t)(b * 2) * HWN + hw];
  Ap[(size_t)(b * 2 + 1) * HWN + hw] = si + lam * p[(size_t)(b * 2 + 1) * HWN + hw];
}

// ---------------- CG init: x=0, r=p=rhs
__global__ void k_cg_init(const float* __restrict__ rhs, float* __restrict__ x,
                          float* __restrict__ r, float* __restrict__ p) {
  int idx = blockIdx.x * 256 + threadIdx.x;
  if (idx >= BN * 2 * HWN) return;
  float v = rhs[idx];
  x[idx] = 0.f; r[idx] = v; p[idx] = v;
}

// ---------------- deterministic two-stage per-batch dot (real part of complex dot)
__global__ void k_dot_partial(const float* __restrict__ A, const float* __restrict__ Bf,
                              float* __restrict__ part) {
  __shared__ float s[256];
  int b = blockIdx.y, blk = blockIdx.x;
  const float* a = A + (size_t)b * 2 * HWN;
  const float* bb = Bf + (size_t)b * 2 * HWN;
  float acc = 0.f;
  for (int i = blk * 256 + threadIdx.x; i < 2 * HWN; i += 64 * 256) acc += a[i] * bb[i];
  s[threadIdx.x] = acc;
  __syncthreads();
  for (int st = 128; st; st >>= 1) {
    if ((int)threadIdx.x < st) s[threadIdx.x] += s[threadIdx.x + st];
    __syncthreads();
  }
  if (threadIdx.x == 0) part[b * 64 + blk] = s[0];
}

__global__ void k_dot_final(const float* __restrict__ part, float* __restrict__ outv) {
  __shared__ float s[64];
  int b = blockIdx.x;
  s[threadIdx.x] = part[b * 64 + threadIdx.x];
  __syncthreads();
  for (int st = 32; st; st >>= 1) {
    if ((int)threadIdx.x < st) s[threadIdx.x] += s[threadIdx.x + st];
    __syncthreads();
  }
  if (threadIdx.x == 0) outv[b] = s[0];
}

// ---------------- CG updates (alpha/beta read from device scalars)
__global__ void k_update1(float* __restrict__ x, float* __restrict__ r,
                          const float* __restrict__ p, const float* __restrict__ Ap,
                          const float* __restrict__ rtr, const float* __restrict__ pAp) {
  int idx = blockIdx.x * 256 + threadIdx.x;
  if (idx >= BN * 2 * HWN) return;
  int b = idx / (2 * HWN);
  float alpha = rtr[b] / pAp[b];
  x[idx] += alpha * p[idx];
  r[idx] -= alpha * Ap[idx];
}

__global__ void k_update2(float* __restrict__ p, const float* __restrict__ r,
                          const float* __restrict__ rtrOld, const float* __restrict__ rtrNew) {
  int idx = blockIdx.x * 256 + threadIdx.x;
  if (idx >= BN * 2 * HWN) return;
  int b = idx / (2 * HWN);
  float beta = rtrNew[b] / rtrOld[b];
  p[idx] = r[idx] + beta * p[idx];
}

extern "C" void kernel_launch(void* const* d_in, const int* in_sizes, int n_in,
                              void* d_out, int out_size, void* d_ws, size_t ws_size,
                              hipStream_t stream) {
  (void)in_sizes; (void)n_in; (void)out_size; (void)ws_size;
  const float* under = (const float*)d_in[0];
  const float* csm   = (const float*)d_in[1];
  const float* mask  = (const float*)d_in[2];
  const float* lamp  = (const float*)d_in[3];
  const float* w_in  = (const float*)d_in[4];
  const float* b_in  = (const float*)d_in[5];
  const float* wmid  = (const float*)d_in[6];
  const float* bmid  = (const float*)d_in[7];
  const float* w_out = (const float*)d_in[8];
  const float* b_out = (const float*)d_in[9];

  char* base = (char*)d_ws;
  size_t off = 0;
  auto carve = [&](size_t bytes) -> char* {
    off = (off + 255) & ~(size_t)255;
    char* ptr = base + off;
    off += bytes;
    return ptr;
  };
  _Float16* actA = (_Float16*)carve((size_t)BN * HWN * NCH * 2);
  _Float16* actB = (_Float16*)carve((size_t)BN * HWN * NCH * 2);
  _Float16* wrep = (_Float16*)carve((size_t)7 * WREP_LAYER * 2);
  float* Wr  = (float*)carve((size_t)HWN * 4);
  float* WiF = (float*)carve((size_t)HWN * 4);
  float* WiI = (float*)carve((size_t)HWN * 4);
  float* rhs = (float*)carve((size_t)BN * 2 * HWN * 4);
  float* xA  = (float*)carve((size_t)BN * 2 * HWN * 4);
  float* xB  = (float*)carve((size_t)BN * 2 * HWN * 4);
  float* rf  = (float*)carve((size_t)BN * 2 * HWN * 4);
  float* pf  = (float*)carve((size_t)BN * 2 * HWN * 4);
  float* Ap  = (float*)carve((size_t)BN * 2 * HWN * 4);
  float* part = (float*)carve((size_t)BN * 64 * 4);
  float* pApS = (float*)carve((size_t)BN * 4);
  float* rtrA = (float*)carve((size_t)BN * 4);
  float* rtrB = (float*)carve((size_t)BN * 4);
  // CG big buffers alias conv activations (temporally disjoint phases)
  float* coil = (float*)actA;   // needs BN*CN*2*HWN*4 = 37.7 MB <= 75.5 MB
  float* tmpK = (float*)actB;

  k_init_dft<<<dim3((HWN + 255) / 256), 256, 0, stream>>>(Wr, WiF, WiI);
  k_repack<<<dim3((7 * WREP_LAYER + 255) / 256), 256, 0, stream>>>(wmid, wrep);

  const float* xcur = under;
  float* xnext = xA;
  for (int layer = 0; layer < NLAYER; ++layer) {
    // ---- denoiser D_w
    k_conv_in<<<dim3((BN * HWN + 255) / 256), 256, 0, stream>>>(xcur, w_in, b_in, actA);
    const _Float16* ain = actA;
    _Float16* aout = actB;
    for (int i = 0; i < 7; ++i) {
      k_conv_mid<<<dim3(WN / 64, HN, BN), 256, 0, stream>>>(
          ain, wrep + (size_t)i * WREP_LAYER, bmid + i * 64, (i % 2 == 0) ? 1 : 0, aout);
      const _Float16* t = ain; ain = aout; aout = (_Float16*)t;
    }
    k_conv_out<<<dim3((BN * HWN + 255) / 256), 256, 0, stream>>>(ain, w_out, b_out, xcur,
                                                                 under, lamp, rhs);
    // ---- CG solve (act buffers free now; coil/tmpK alias them)
    float* xcg = xnext;
    k_cg_init<<<dim3((BN * 2 * HWN + 255) / 256), 256, 0, stream>>>(rhs, xcg, rf, pf);
    k_dot_partial<<<dim3(64, BN), 256, 0, stream>>>(rf, rf, part);
    k_dot_final<<<dim3(BN), 64, 0, stream>>>(part, rtrA);
    float* rtrCur = rtrA;
    float* rtrNew = rtrB;
    for (int it = 0; it < CGITER; ++it) {
      k_coil_expand<<<dim3((BN * CN * HWN + 255) / 256), 256, 0, stream>>>(pf, csm, coil);
      // fft2 = two transposed GEMM stages against the symmetric forward table
      k_cgemm<<<dim3(6, 24, BN * CN), 128, 0, stream>>>(coil, coil + HWN, 2LL * HWN,
              Wr, WiF, tmpK, tmpK + HWN, 2LL * HWN, 1.0f);
      k_cgemm<<<dim3(6, 24, BN * CN), 128, 0, stream>>>(tmpK, tmpK + HWN, 2LL * HWN,
              Wr, WiF, coil, coil + HWN, 2LL * HWN, 1.0f);
      k_mask_mul<<<dim3((BN * CN * 2 * HWN + 255) / 256), 256, 0, stream>>>(coil, mask);
      // ifft2 = two transposed GEMM stages against the conjugate table, 1/(H*W) on stage 2
      k_cgemm<<<dim3(6, 24, BN * CN), 128, 0, stream>>>(coil, coil + HWN, 2LL * HWN,
              Wr, WiI, tmpK, tmpK + HWN, 2LL * HWN, 1.0f);
      k_cgemm<<<dim3(6, 24, BN * CN), 128, 0, stream>>>(tmpK, tmpK + HWN, 2LL * HWN,
              Wr, WiI, coil, coil + HWN, 2LL * HWN, 1.0f / (float)HWN);
      k_coil_reduce<<<dim3((BN * HWN + 255) / 256), 256, 0, stream>>>(coil, csm, pf, lamp, Ap);
      k_dot_partial<<<dim3(64, BN), 256, 0, stream>>>(pf, Ap, part);
      k_dot_final<<<dim3(BN), 64, 0, stream>>>(part, pApS);
      k_update1<<<dim3((BN * 2 * HWN + 255) / 256), 256, 0, stream>>>(xcg, rf, pf, Ap,
                                                                      rtrCur, pApS);
      k_dot_partial<<<dim3(64, BN), 256, 0, stream>>>(rf, rf, part);
      k_dot_final<<<dim3(BN), 64, 0, stream>>>(part, rtrNew);
      k_update2<<<dim3((BN * 2 * HWN + 255) / 256), 256, 0, stream>>>(pf, rf, rtrCur, rtrNew);
      float* t = rtrCur; rtrCur = rtrNew; rtrNew = t;
    }
    xcur = xcg;
    xnext = (xcg == xA) ? xB : xA;
  }
  hipMemcpyAsync(d_out, (const void*)xcur, (size_t)BN * 2 * HWN * 4,
                 hipMemcpyDeviceToDevice, stream);
}
